// BridgeNet_37855841747274
// MI455X (gfx1250) — matrix operand order repaired and checked by reading, not compiled, not run on hardware
//
#include <hip/hip_runtime.h>
#include <hip/hip_bf16.h>
#include <cstdint>
#include <cstddef>

// ---------------------------------------------------------------------------
// BridgeNet block for MI455X (gfx1250, wave32).
// GEMMs use v_wmma_f32_16x16x32_f16 (f32 accumulate). BN is training-mode ->
// conv biases cancel exactly; att_b is softmax-invariant -> dropped exactly.
// ---------------------------------------------------------------------------

typedef __attribute__((ext_vector_type(16))) _Float16 v16h;
typedef __attribute__((ext_vector_type(8)))  float    v8f;

static constexpr int kB   = 4;
static constexpr int kN   = 4096;
static constexpr int kP   = kB * kN;      // 16384 points
static constexpr int kNS  = 16;           // neighbors
static constexpr int kRG  = kP * kNS;     // 262144 (point,neighbor) rows
static constexpr int kC   = 128;
static constexpr int kSC  = 32;
static constexpr float kEPS = 1e-5f;

// ---------------------------------------------------------------------------
// f32 -> f16 weight conversion
// ---------------------------------------------------------------------------
__global__ void cvt_f16_kernel(const float* __restrict__ src,
                               _Float16* __restrict__ dst, int n) {
  int i = blockIdx.x * 256 + threadIdx.x;
  if (i < n) dst[i] = (_Float16)src[i];
}

// ---------------------------------------------------------------------------
// KNN: one thread per query point, candidate tiles staged in LDS (12 KB).
// Sorted-ascending insertion keeps ties ordered by lower index (matches
// top_k stability); self excluded explicitly. Writes global neighbor row ids.
// ---------------------------------------------------------------------------
__global__ __launch_bounds__(256) void knn_kernel(const float* __restrict__ points,
                                                  int* __restrict__ idxg) {
  __shared__ float sx[1024], sy[1024], sz[1024];
  const int blocksPerBatch = kN / 256;
  const int b  = blockIdx.x / blocksPerBatch;
  const int q  = (blockIdx.x % blocksPerBatch) * 256 + threadIdx.x; // in-batch
  const float* pb = points + (size_t)b * kN * 3;
  const float qx = pb[q * 3 + 0], qy = pb[q * 3 + 1], qz = pb[q * 3 + 2];

  float bd[kNS];
  int   bi[kNS];
#pragma unroll
  for (int t = 0; t < kNS; ++t) { bd[t] = 3.0e38f; bi[t] = 0; }

  for (int t0 = 0; t0 < kN; t0 += 1024) {
    for (int u = threadIdx.x; u < 1024; u += 256) {
      sx[u] = pb[(t0 + u) * 3 + 0];
      sy[u] = pb[(t0 + u) * 3 + 1];
      sz[u] = pb[(t0 + u) * 3 + 2];
    }
    __syncthreads();
    for (int j = 0; j < 1024; ++j) {
      const int jj = t0 + j;
      const float dx = qx - sx[j], dy = qy - sy[j], dz = qz - sz[j];
      const float d2 = dx * dx + dy * dy + dz * dz;
      if (jj != q && d2 < bd[kNS - 1]) {
        bd[kNS - 1] = d2; bi[kNS - 1] = jj;
#pragma unroll
        for (int t = kNS - 1; t > 0; --t) {
          if (bd[t] < bd[t - 1]) {
            float td = bd[t]; bd[t] = bd[t - 1]; bd[t - 1] = td;
            int   ti = bi[t]; bi[t] = bi[t - 1]; bi[t - 1] = ti;
          }
        }
      }
    }
    __syncthreads();
  }
  const int rowBase = (b * kN + q) * kNS;
#pragma unroll
  for (int s = 0; s < kNS; ++s) idxg[rowBase + s] = b * kN + bi[s];
}

// ---------------------------------------------------------------------------
// Geo encoding: per (row, out-channel): [xi,xj,xi-xj,|xi-xj|] . geo_w[c,:]
// (geo bias cancels under training BN)
// ---------------------------------------------------------------------------
__global__ void geo_kernel(const float* __restrict__ points,
                           const int* __restrict__ idxg,
                           const float* __restrict__ geo_w,
                           float* __restrict__ geo_raw) {
  const int gi  = blockIdx.x * 256 + threadIdx.x;   // over kRG * kSC
  const int row = gi >> 5;
  const int c   = gi & 31;
  const int p   = row >> 4;
  const int jg  = idxg[row];
  const float xi0 = points[p * 3 + 0], xi1 = points[p * 3 + 1], xi2 = points[p * 3 + 2];
  const float xj0 = points[jg * 3 + 0], xj1 = points[jg * 3 + 1], xj2 = points[jg * 3 + 2];
  const float r0 = xi0 - xj0, r1 = xi1 - xj1, r2 = xi2 - xj2;
  const float dist = sqrtf(r0 * r0 + r1 * r1 + r2 * r2);
  const float* w = geo_w + c * 10;
  float acc = xi0 * w[0] + xi1 * w[1] + xi2 * w[2]
            + xj0 * w[3] + xj1 * w[4] + xj2 * w[5]
            + r0 * w[6] + r1 * w[7] + r2 * w[8] + dist * w[9];
  geo_raw[gi] = acc;
}

// ---------------------------------------------------------------------------
// Per-channel training-BN stats (deterministic, one workgroup per channel).
// Emits folded affine: scale = g*rsqrt(var+eps), shift = bt - mean*scale.
// ---------------------------------------------------------------------------
__global__ void colstats_kernel(const float* __restrict__ X, int R, int Cch,
                                const float* __restrict__ g,
                                const float* __restrict__ bt,
                                float* __restrict__ scale,
                                float* __restrict__ shift) {
  __shared__ float s1[256], s2[256];
  const int c = blockIdx.x;
  float a = 0.f, b2 = 0.f;
  for (int r = threadIdx.x; r < R; r += 256) {
    const float v = X[(size_t)r * Cch + c];
    a += v; b2 += v * v;
  }
  s1[threadIdx.x] = a; s2[threadIdx.x] = b2;
  __syncthreads();
  for (int s = 128; s > 0; s >>= 1) {
    if (threadIdx.x < s) { s1[threadIdx.x] += s1[threadIdx.x + s]; s2[threadIdx.x] += s2[threadIdx.x + s]; }
    __syncthreads();
  }
  if (threadIdx.x == 0) {
    const float invR = 1.f / (float)R;
    const float mean = s1[0] * invR;
    const float var  = s2[0] * invR - mean * mean;
    const float istd = rsqrtf(var + kEPS);
    const float sc   = g[c] * istd;
    scale[c] = sc;
    shift[c] = bt[c] - mean * sc;
  }
}

// ---------------------------------------------------------------------------
// Generic WMMA GEMM: D[M,NC] = A[M,K](f32->f16) x W[NC,K](f16)^T, f32 accum.
// One wave -> one 16x16 tile; 8 waves/block; K stepped by 32.
// ---------------------------------------------------------------------------
__global__ __launch_bounds__(256) void gemm_wmma_kernel(
    const float* __restrict__ A, const _Float16* __restrict__ W,
    float* __restrict__ D, int K, int NC) {
  const int lane = threadIdx.x & 31;
  const int wave = threadIdx.x >> 5;
  const int nTilesN = NC >> 4;
  const int tile = blockIdx.x * 8 + wave;
  const int rowBase = (tile / nTilesN) << 4;
  const int n0      = (tile % nTilesN) << 4;
  const int l = lane & 15, h = lane >> 4;

  const float*    ap = A + (size_t)(rowBase + l) * K;
  const _Float16* wp = W + (size_t)(n0 + l) * K + h * 16;

  v8f acc = {0.f, 0.f, 0.f, 0.f, 0.f, 0.f, 0.f, 0.f};
  for (int k0 = 0; k0 < K; k0 += 32) {
    v16h a, b;
#pragma unroll
    for (int e = 0; e < 8; ++e) {
      a[e]     = (_Float16)ap[k0 + h * 16 + e];       // A: K = k0 + h*16 + e
      a[e + 8] = (_Float16)ap[k0 + h * 16 + 8 + e];   // A: K = k0 + h*16 + 8 + e
    }
#pragma unroll
    for (int e = 0; e < 16; ++e) b[e] = wp[k0 + e];   // B: K = k0 + h*16 + e
    acc = __builtin_amdgcn_wmma_f32_16x16x32_f16(false, a, false, b,
                                                 (short)0, acc, false, false);
  }
  float* dp = D + (size_t)rowBase * NC + n0 + l;
#pragma unroll
  for (int r = 0; r < 8; ++r) dp[(size_t)(r + h * 8) * NC] = acc[r];
}

// ---------------------------------------------------------------------------
// grp WMMA GEMM with fused gather + BN + ReLU on the A operand:
// A[row, 0:32]  = relu(BN(geo_raw[row]))
// A[row, 32:64] = relu(BN(sub_raw[idxg[row]]))   (neighbor gather)
// K = 64, NC = 32, M = kRG.
// ---------------------------------------------------------------------------
__global__ __launch_bounds__(256) void gemm_grp_wmma_kernel(
    const float* __restrict__ geo_raw, const float* __restrict__ gs, const float* __restrict__ gsh,
    const float* __restrict__ sub_raw, const float* __restrict__ ss, const float* __restrict__ ssh,
    const int* __restrict__ idxg, const _Float16* __restrict__ W,
    float* __restrict__ D) {
  const int lane = threadIdx.x & 31;
  const int wave = threadIdx.x >> 5;
  const int tile = blockIdx.x * 8 + wave;       // nTilesN = 2
  const int rowBase = (tile >> 1) << 4;
  const int n0      = (tile & 1) << 4;
  const int l = lane & 15, h = lane >> 4;

  const int row = rowBase + l;
  const int jg  = idxg[row];
  const float* gp = geo_raw + (size_t)row * kSC;
  const float* sp = sub_raw + (size_t)jg * kSC;
  const _Float16* wp = W + (size_t)(n0 + l) * 64 + h * 16;

  v8f acc = {0.f, 0.f, 0.f, 0.f, 0.f, 0.f, 0.f, 0.f};
  v16h a, b;

  // k-step 0: geo channels 0..31
#pragma unroll
  for (int e = 0; e < 8; ++e) {
    int k = h * 16 + e;
    a[e] = (_Float16)fmaxf(gp[k] * gs[k] + gsh[k], 0.f);
    k = h * 16 + 8 + e;
    a[e + 8] = (_Float16)fmaxf(gp[k] * gs[k] + gsh[k], 0.f);
  }
#pragma unroll
  for (int e = 0; e < 16; ++e) b[e] = wp[e];
  acc = __builtin_amdgcn_wmma_f32_16x16x32_f16(false, a, false, b, (short)0, acc, false, false);

  // k-step 1: gathered sub channels 0..31
#pragma unroll
  for (int e = 0; e < 8; ++e) {
    int k = h * 16 + e;
    a[e] = (_Float16)fmaxf(sp[k] * ss[k] + ssh[k], 0.f);
    k = h * 16 + 8 + e;
    a[e + 8] = (_Float16)fmaxf(sp[k] * ss[k] + ssh[k], 0.f);
  }
#pragma unroll
  for (int e = 0; e < 16; ++e) b[e] = wp[32 + e];
  acc = __builtin_amdgcn_wmma_f32_16x16x32_f16(false, a, false, b, (short)0, acc, false, false);

  float* dp = D + (size_t)rowBase * kSC + n0 + l;
#pragma unroll
  for (int r = 0; r < 8; ++r) dp[(size_t)(r + h * 8) * kSC] = acc[r];
}

// ---------------------------------------------------------------------------
// Attention epilogue: one wave per point (lane == channel, wave32 == SC).
// h = relu(BN(grp_raw)); score = softmax_ch(sum_k h*att_w[k]);
// out = maxpool_k(h) * (1 + score).  (att_b is softmax-invariant -> dropped)
// ---------------------------------------------------------------------------
__device__ inline float wred_max(float v) {
#pragma unroll
  for (int m = 16; m >= 1; m >>= 1) v = fmaxf(v, __shfl_xor(v, m, 32));
  return v;
}
__device__ inline float wred_sum(float v) {
#pragma unroll
  for (int m = 16; m >= 1; m >>= 1) v += __shfl_xor(v, m, 32);
  return v;
}

__global__ __launch_bounds__(256) void attn_kernel(
    const float* __restrict__ grp_raw,
    const float* __restrict__ gs, const float* __restrict__ gsh,
    const float* __restrict__ att_w, float* __restrict__ outb) {
  const int wave = threadIdx.x >> 5;
  const int c    = threadIdx.x & 31;
  const int p    = blockIdx.x * 8 + wave;
  const float sc = gs[c], sh = gsh[c];
  const float* base = grp_raw + ((size_t)p * kNS) * kSC + c;
  float att = 0.f, mx = -3.0e38f;
#pragma unroll
  for (int kk = 0; kk < kNS; ++kk) {
    const float hx = fmaxf(base[kk * kSC] * sc + sh, 0.f);
    att = fmaf(hx, att_w[kk], att);
    mx  = fmaxf(mx, hx);
  }
  const float m = wred_max(att);
  const float e = expf(att - m);
  const float s = wred_sum(e);
  outb[(size_t)p * kSC + c] = mx * (1.f + e / s);
}

// ---------------------------------------------------------------------------
// feat_next = relu(BN(up_raw)) + relu(BN(res_raw))
// ---------------------------------------------------------------------------
__global__ void combine_kernel(const float* __restrict__ up_raw,
                               const float* __restrict__ us, const float* __restrict__ ush,
                               const float* __restrict__ res_raw,
                               const float* __restrict__ rs, const float* __restrict__ rsh,
                               float* __restrict__ out) {
  const int gi = blockIdx.x * 256 + threadIdx.x;  // over kP * kC
  const int c  = gi & (kC - 1);
  out[gi] = fmaxf(up_raw[gi] * us[c] + ush[c], 0.f)
          + fmaxf(res_raw[gi] * rs[c] + rsh[c], 0.f);
}

// ---------------------------------------------------------------------------
// Host orchestration
// ---------------------------------------------------------------------------
extern "C" void kernel_launch(void* const* d_in, const int* in_sizes, int n_in,
                              void* d_out, int out_size, void* d_ws, size_t ws_size,
                              hipStream_t stream) {
  (void)in_sizes; (void)n_in; (void)out_size; (void)ws_size;
  const float* points   = (const float*)d_in[0];
  const float* features = (const float*)d_in[1];
  const float* geo_w    = (const float*)d_in[2];
  const float* geo_g    = (const float*)d_in[4];
  const float* geo_bt   = (const float*)d_in[5];
  const float* sub_w    = (const float*)d_in[6];
  const float* sub_g    = (const float*)d_in[8];
  const float* sub_bt   = (const float*)d_in[9];
  const float* grp_w    = (const float*)d_in[10];
  const float* grp_g    = (const float*)d_in[12];
  const float* grp_bt   = (const float*)d_in[13];
  const float* att_w    = (const float*)d_in[14];
  const float* up_w     = (const float*)d_in[16];
  const float* up_g     = (const float*)d_in[18];
  const float* up_bt    = (const float*)d_in[19];
  const float* res_w    = (const float*)d_in[20];
  const float* res_g    = (const float*)d_in[22];
  const float* res_bt   = (const float*)d_in[23];

  // ---- workspace carve-up (up_raw/res_raw alias grp_raw: grp is dead by then)
  char* ws = (char*)d_ws;
  size_t off = 0;
  auto carve = [&](size_t bytes) -> char* {
    char* p = ws + off;
    off = (off + bytes + 255) & ~(size_t)255;
    return p;
  };
  int*   idxg    = (int*)  carve((size_t)kRG * 4);
  float* geo_raw = (float*)carve((size_t)kRG * kSC * 4);
  float* sub_raw = (float*)carve((size_t)kP * kSC * 4);
  float* grp_raw = (float*)carve((size_t)kRG * kSC * 4);   // 8.4M floats
  float* up_raw  = grp_raw;                                 // alias (2.1M floats)
  float* res_raw = grp_raw + (size_t)kP * kC;               // alias (2.1M floats)
  float* outb    = (float*)carve((size_t)kP * kSC * 4);
  float* featB   = (float*)carve((size_t)kP * kC * 4);
  _Float16* sub16 = (_Float16*)carve((size_t)2 * kSC * kC * 2);
  _Float16* grp16 = (_Float16*)carve((size_t)2 * kSC * 64 * 2);
  _Float16* up16  = (_Float16*)carve((size_t)2 * kC * kSC * 2);
  _Float16* res16 = (_Float16*)carve((size_t)2 * kC * kC * 2);
  float* gscale = (float*)carve(kSC * 4);
  float* gshift = (float*)carve(kSC * 4);
  float* sscale = (float*)carve(kSC * 4);
  float* sshift = (float*)carve(kSC * 4);
  float* pscale = (float*)carve(kSC * 4);   // grp stats
  float* pshift = (float*)carve(kSC * 4);
  float* uscale = (float*)carve(kC * 4);
  float* ushift = (float*)carve(kC * 4);
  float* rscale = (float*)carve(kC * 4);
  float* rshift = (float*)carve(kC * 4);

  // ---- weight conversion to f16
  auto cvt = [&](const float* s, _Float16* d, int n) {
    cvt_f16_kernel<<<(n + 255) / 256, 256, 0, stream>>>(s, d, n);
  };
  cvt(sub_w, sub16, 2 * kSC * kC);
  cvt(grp_w, grp16, 2 * kSC * 64);
  cvt(up_w,  up16,  2 * kC * kSC);
  cvt(res_w, res16, 2 * kC * kC);

  // ---- KNN + geo encoding + geo BN stats
  knn_kernel<<<kB * (kN / 256), 256, 0, stream>>>(points, idxg);
  geo_kernel<<<(kRG * kSC) / 256, 256, 0, stream>>>(points, idxg, geo_w, geo_raw);
  colstats_kernel<<<kSC, 256, 0, stream>>>(geo_raw, kRG, kSC, geo_g, geo_bt, gscale, gshift);

  const float* featCur = features;
  for (int i = 0; i < 2; ++i) {
    // sub: [P,128] x [32,128]^T -> [P,32]
    gemm_wmma_kernel<<<(kP / 16) * (kSC / 16) / 8, 256, 0, stream>>>(
        featCur, sub16 + (size_t)i * kSC * kC, sub_raw, kC, kSC);
    colstats_kernel<<<kSC, 256, 0, stream>>>(sub_raw, kP, kSC,
                                             sub_g + i * kSC, sub_bt + i * kSC, sscale, sshift);
    // grp: fused gather+BN+relu A, [RG,64] x [32,64]^T -> [RG,32]
    gemm_grp_wmma_kernel<<<(kRG / 16) * 2 / 8, 256, 0, stream>>>(
        geo_raw, gscale, gshift, sub_raw, sscale, sshift, idxg,
        grp16 + (size_t)i * kSC * 64, grp_raw);
    colstats_kernel<<<kSC, 256, 0, stream>>>(grp_raw, kRG, kSC,
                                             grp_g + i * kSC, grp_bt + i * kSC, pscale, pshift);
    // attention + maxpool epilogue -> outb [P,32]
    attn_kernel<<<kP / 8, 256, 0, stream>>>(grp_raw, pscale, pshift,
                                            att_w + i * kNS, outb);
    // up: [P,32] x [128,32]^T -> [P,128]   (up_raw aliases grp_raw; grp is dead)
    gemm_wmma_kernel<<<(kP / 16) * (kC / 16) / 8, 256, 0, stream>>>(
        outb, up16 + (size_t)i * kC * kSC, up_raw, kSC, kC);
    colstats_kernel<<<kC, 256, 0, stream>>>(up_raw, kP, kC,
                                            up_g + i * kC, up_bt + i * kC, uscale, ushift);
    // res: [P,128] x [128,128]^T -> [P,128]
    gemm_wmma_kernel<<<(kP / 16) * (kC / 16) / 8, 256, 0, stream>>>(
        featCur, res16 + (size_t)i * kC * kC, res_raw, kC, kC);
    colstats_kernel<<<kC, 256, 0, stream>>>(res_raw, kP, kC,
                                            res_g + i * kC, res_bt + i * kC, rscale, rshift);
    // feat = relu(BN(up)) + relu(BN(res))
    float* dst = (i == 0) ? featB : (float*)d_out;
    combine_kernel<<<(kP * kC) / 256, 256, 0, stream>>>(
        up_raw, uscale, ushift, res_raw, rscale, rshift, dst);
    featCur = featB;
  }
}
